// StackedTLSTM_43980465111646
// MI455X (gfx1250) — matrix operand order repaired and checked by reading, not compiled, not run on hardware
//
#include <hip/hip_runtime.h>

// ---------------- problem constants ----------------
#define BB 256
#define SS 128
#define II 256
#define HH 1024
#define LL 2
#define ECONST 2.7183f
#define EPSC 1e-6f

typedef __bf16 bf16;
typedef __attribute__((ext_vector_type(16))) __bf16 v16bf;
typedef __attribute__((ext_vector_type(8)))  float  v8f;
typedef unsigned int u32x4 __attribute__((ext_vector_type(4)));
typedef int          i32x4 __attribute__((ext_vector_type(4)));
typedef int          i32x8 __attribute__((ext_vector_type(8)));

union FragU { uint4 u[2]; v16bf v; };

// gfx1250 async copy: memory -> LDS, tracked by ASYNCcnt (no VGPR round-trip)
__device__ __forceinline__ void async_copy_b128(unsigned lds_off, unsigned long long gaddr)
{
    asm volatile("global_load_async_to_lds_b128 %0, %1, off"
                 :: "v"(lds_off), "v"(gaddr)
                 : "memory");
}
__device__ __forceinline__ unsigned lds_off_of(const void* p)
{
    return (unsigned)(unsigned long long)p;   // flat shared ptr: low 32 bits = LDS offset
}

// gfx1250 Tensor Data Mover: DMA a 2D tile (tile1_rows x tile0_elems of 2-byte
// elements, row stride in elements) from global memory into LDS. One instruction
// per tile, tracked with TENSORcnt. Descriptor per CDNA5 ISA.
__device__ __forceinline__ void tdm_load_2d_b16(unsigned lds_addr, unsigned long long gaddr,
                                                unsigned tile0_elems, unsigned tile1_rows,
                                                unsigned long long row_stride_elems)
{
    u32x4 g0;
    g0[0] = 1u;                                              // count=1, user mode
    g0[1] = lds_addr;                                        // LDS byte address
    g0[2] = (unsigned)(gaddr & 0xffffffffu);                 // global_addr[31:0]
    g0[3] = (unsigned)((gaddr >> 32) & 0x1ffffffu) | (2u << 30); // addr[56:32], type=2
    const unsigned long long d0   = row_stride_elems;        // tensor_dim0 (OOB bound)
    const unsigned long long dim1 = 1u << 20;                // generous tensor_dim1
    i32x8 g1;
    g1[0] = (int)(1u << 16);                                 // wg_mask=0, data_size=2B
    g1[1] = (int)((d0 & 0xffffu) << 16);                     // tensor_dim0[15:0]
    g1[2] = (int)(((d0 >> 16) & 0xffffu) | ((dim1 & 0xffffu) << 16));
    g1[3] = (int)(((dim1 >> 16) & 0xffffu) | (tile0_elems << 16)); // tile_dim0
    g1[4] = (int)(tile1_rows & 0xffffu);                     // tile_dim1, tile_dim2=0
    g1[5] = (int)(row_stride_elems & 0xffffffffu);           // tensor_dim0_stride lo
    g1[6] = (int)((row_stride_elems >> 32) & 0xffffu);       // stride hi, dim1_stride=0
    g1[7] = 0;
    i32x4 g2; g2[0] = 1; g2[1] = 1; g2[2] = 0; g2[3] = 0;    // tensor_dim2/3 = 1
    i32x4 g3; g3[0] = 0; g3[1] = 0; g3[2] = 0; g3[3] = 0;
#if __clang_major__ >= 23
    i32x8 g4; g4[0]=0; g4[1]=0; g4[2]=0; g4[3]=0; g4[4]=0; g4[5]=0; g4[6]=0; g4[7]=0;
    __builtin_amdgcn_tensor_load_to_lds(g0, g1, g2, g3, g4, 0);
#else
    __builtin_amdgcn_tensor_load_to_lds(g0, g1, g2, g3, 0);
#endif
}

// ---------------- WMMA GEMM ----------------
// C[M x N] (f32, row-major, ldc) = A[M x K] (bf16) * Bt^T, Bt stored K-major
// (Bt[n][k], ldb elements per row). A split at K0 between A0/A1 sources.
// WG = 128 threads (4 waves), tile 64(M) x 128(N), TK=32, double-buffered LDS.
// B tile (8 KB) staged by one TDM tensor_load_to_lds issued by wave 0
// (TENSORcnt); A tile (4 KB) staged by per-thread async b128 copies (ASYNCcnt).
// Each wave computes 32x64 via 2x4 v_wmma_f32_16x16x32_bf16 accumulators.
#define TM 64
#define TN 128
#define TK 32

__launch_bounds__(128)
__global__ void wmma_gemm_bf16(const bf16* __restrict__ A0, long lda0,
                               const bf16* __restrict__ A1, long lda1,
                               int K0,
                               const bf16* __restrict__ Bt, long ldb,
                               float* __restrict__ C, long ldc,
                               int N, int K)
{
    __shared__ __align__(16) bf16 sA[2][TM][TK];   // 2 x 4 KB
    __shared__ __align__(16) bf16 sB[2][TN][TK];   // 2 x 8 KB

    const int tid   = threadIdx.x;
    const int lane  = tid & 31;
    const int wave  = tid >> 5;
    const int wm    = (wave >> 1) << 5;   // 0 or 32
    const int wn    = (wave & 1) << 6;    // 0 or 64
    const int m0    = blockIdx.y * TM;
    const int n0    = blockIdx.x * TN;
    const int lhalf = lane >> 4;          // 0/1
    const int l15   = lane & 15;

    v8f acc[2][4];
#pragma unroll
    for (int i = 0; i < 2; ++i)
#pragma unroll
        for (int j = 0; j < 4; ++j)
#pragma unroll
            for (int r = 0; r < 8; ++r) acc[i][j][r] = 0.0f;

    // per-thread A staging coordinates
    const int ar   = tid >> 1;            // A row 0..63
    const int aseg = (tid & 1) << 4;      // A k-seg 0/16

    // A tile: 2 async b128 per thread into buffer `buf`
    auto stage_A = [&](int buf, int kc) {
        const bf16* Ap; long lda; int kk;
        if (kc < K0) { Ap = A0; lda = lda0; kk = kc; }
        else         { Ap = A1; lda = lda1; kk = kc - K0; }
        const unsigned long long asrc =
            (unsigned long long)(Ap + (long)(m0 + ar) * lda + kk + aseg);
        const unsigned aoff = lds_off_of(&sA[buf][ar][aseg]);
        async_copy_b128(aoff,      asrc);
        async_copy_b128(aoff + 16, asrc + 16);
    };
    // B tile: single TDM descriptor (executed by wave 0 only)
    auto stage_B = [&](int buf, int kc) {
        const unsigned long long bsrc =
            (unsigned long long)(Bt + (long)n0 * ldb + kc);
        tdm_load_2d_b16(lds_off_of(&sB[buf][0][0]), bsrc,
                        TK, TN, (unsigned long long)ldb);
    };

    stage_A(0, 0);
    if (tid < 32) stage_B(0, 0);

    int buf = 0;
    for (int kc = 0; kc < K; kc += TK, buf ^= 1) {
        if (kc + TK < K) {
            stage_A(buf ^ 1, kc + TK);
            if (tid < 32) stage_B(buf ^ 1, kc + TK);
            // 2 new async + 1 new tensor op in flight; waiting to those counts
            // guarantees the previous tile's transfers completed (in-order).
            asm volatile("s_wait_asynccnt 0x2" ::: "memory");
            if (tid < 32) __builtin_amdgcn_s_wait_tensorcnt((short)1);
        } else {
            asm volatile("s_wait_asynccnt 0x0" ::: "memory");
            if (tid < 32) __builtin_amdgcn_s_wait_tensorcnt((short)0);
        }
        __syncthreads();

        // ---- fragments per documented wave32 layouts ----
        v16bf afrag[2], bfrag[4];
#pragma unroll
        for (int mi = 0; mi < 2; ++mi) {
            const bf16* base = &sA[buf][wm + mi * 16 + l15][0];
            FragU f;
            f.u[0] = *(const uint4*)(base + (lhalf << 3));        // K = kb..kb+7
            f.u[1] = *(const uint4*)(base + 16 + (lhalf << 3));   // K = 16+kb..
            afrag[mi] = f.v;
        }
#pragma unroll
        for (int ni = 0; ni < 4; ++ni) {
            const bf16* base = &sB[buf][wn + ni * 16 + l15][0];
            FragU f;
            f.u[0] = *(const uint4*)(base + (lhalf << 3));
            f.u[1] = *(const uint4*)(base + 16 + (lhalf << 3));
            bfrag[ni] = f.v;
        }
#pragma unroll
        for (int mi = 0; mi < 2; ++mi)
#pragma unroll
            for (int ni = 0; ni < 4; ++ni)
                acc[mi][ni] = __builtin_amdgcn_wmma_f32_16x16x32_bf16(
                    false, afrag[mi], false, bfrag[ni],
                    (short)0, acc[mi][ni], false, false);
        __syncthreads();   // protect buf from being restaged while in use
    }

    // ---- epilogue: C/D layout -> VGPR r holds M = r + 8*(lane>=16) ----
#pragma unroll
    for (int mi = 0; mi < 2; ++mi) {
        const int row0 = m0 + wm + mi * 16 + (lhalf << 3);
#pragma unroll
        for (int ni = 0; ni < 4; ++ni) {
            const int col = n0 + wn + ni * 16 + l15;
#pragma unroll
            for (int r = 0; r < 8; ++r)
                C[(long)(row0 + r) * ldc + col] = acc[mi][ni][r];
        }
    }
}

// ---------------- helpers ----------------
__global__ void compute_T_kernel(const float* __restrict__ ts,
                                 float* __restrict__ T, long n)
{
    long i = (long)blockIdx.x * 256 + threadIdx.x;
    if (i < n) T[i] = 1.0f / logf(ts[i] + ECONST + EPSC);
}

__global__ void f32_to_bf16_kernel(const float* __restrict__ src,
                                   bf16* __restrict__ dst, long n)
{
    long stride = (long)gridDim.x * 256;
    for (long i = (long)blockIdx.x * 256 + threadIdx.x; i < n; i += stride)
        dst[i] = (bf16)src[i];
}

// Build Wcat_t[n][k] (K-major): n = g*H + row, k<in -> W_g[row][k], else U_g[row][k-in]
__global__ void pack_gates_kernel(const float* __restrict__ Wi, const float* __restrict__ Ui,
                                  const float* __restrict__ Wf, const float* __restrict__ Uf,
                                  const float* __restrict__ Wo, const float* __restrict__ Uo,
                                  const float* __restrict__ Wc, const float* __restrict__ Uc,
                                  bf16* __restrict__ out, int in_dim)
{
    const long K = in_dim + HH;
    const long total = 4L * HH * K;
    const long stride = (long)gridDim.x * 256;
    for (long idx = (long)blockIdx.x * 256 + threadIdx.x; idx < total; idx += stride) {
        const long n = idx / K;
        const int  k = (int)(idx - n * K);
        const int  g = (int)(n >> 10);
        const int  row = (int)(n & 1023);
        const float* W; const float* U;
        switch (g) {
            case 0:  W = Wi; U = Ui; break;
            case 1:  W = Wf; U = Uf; break;
            case 2:  W = Wo; U = Uo; break;
            default: W = Wc; U = Uc; break;
        }
        const float v = (k < in_dim) ? W[(long)row * in_dim + k]
                                     : U[(long)row * HH + (k - in_dim)];
        out[idx] = (bf16)v;
    }
}

// Fused T-LSTM pointwise step.
__global__ void tlstm_point_kernel(const float* __restrict__ G,   // [B,4H] gate pre-acts
                                   const float* __restrict__ Dp,  // [B,H] decay pre-act
                                   const float* __restrict__ bi, const float* __restrict__ bfb,
                                   const float* __restrict__ bo, const float* __restrict__ bc,
                                   const float* __restrict__ bd,
                                   const float* __restrict__ Tcol, // = Tbuf + t; index b*S
                                   float* __restrict__ h, float* __restrict__ c,
                                   bf16* __restrict__ hbf, bf16* __restrict__ cbf,
                                   float* __restrict__ outf,  // nullable: d_out + t*H
                                   bf16*  __restrict__ outbf) // nullable: x1bf + t*H
{
    const int idx = blockIdx.x * 256 + threadIdx.x;   // < B*H
    const int b = idx >> 10;
    const int j = idx & 1023;

    float cv  = c[idx];
    const float cst = tanhf(Dp[idx] + bd[j]);
    const float Tt  = Tcol[(long)b * SS];
    cv = cv - cst + Tt * cst;

    const float* g = G + (long)b * 4096;
    const float ig = 1.0f / (1.0f + __expf(-(g[j]          + bi[j])));
    const float fg = 1.0f / (1.0f + __expf(-(g[1024 + j]   + bfb[j])));
    const float og = 1.0f / (1.0f + __expf(-(g[2048 + j]   + bo[j])));
    const float gg = tanhf(g[3072 + j] + bc[j]);

    cv = fg * cv + ig * gg;
    const float hv = og * tanhf(cv);

    c[idx] = cv;  h[idx] = hv;
    cbf[idx] = (bf16)cv;  hbf[idx] = (bf16)hv;

    const long o = (long)b * (SS * HH) + j;
    if (outf)  outf[o]  = hv;
    if (outbf) outbf[o] = (bf16)hv;
}

// ---------------- host ----------------
extern "C" void kernel_launch(void* const* d_in, const int* in_sizes, int n_in,
                              void* d_out, int out_size, void* d_ws, size_t ws_size,
                              hipStream_t stream)
{
    (void)out_size; (void)ws_size;
    // canonical param order: Wi bi Ui Wf bf Uf Wo bo Uo Wc bc Uc Wd bd
    static const int mapIns[14]  = {0,1,2,3,4,5,6,7,8,9,10,11,12,13};
    static const int mapSort[14] = {7,12,2,6,11,1,8,13,3,4,9,0,5,10}; // sorted-key pytree order

    const bool insertion = (n_in >= 2 && in_sizes[1] == BB * SS);
    const float* x_in = (const float*)d_in[0];
    const float* ts   = (const float*)d_in[insertion ? 1 : (n_in - 1)];
    const int pbase   = insertion ? 2 : 1;
    const int* mp     = insertion ? mapIns : mapSort;

    const float* P[LL][14];
    for (int l = 0; l < LL; ++l)
        for (int i = 0; i < 14; ++i)
            P[l][i] = (const float*)d_in[pbase + 14 * l + mp[i]];
    enum { pWi=0,pbi,pUi,pWf,pbf,pUf,pWo,pbo,pUo,pWc,pbc,pUc,pWd,pbd };

    // ---- workspace carve-up (256B aligned) ----
    char* ws = (char*)d_ws;
    size_t off = 0;
    auto take = [&](size_t bytes) { size_t o = off; off = (off + bytes + 255) & ~(size_t)255; return o; };
    float* Tbuf   = (float*)(ws + take((size_t)BB * SS * 4));
    bf16*  x0bf   = (bf16*) (ws + take((size_t)BB * SS * II * 2));
    bf16*  x1bf   = (bf16*) (ws + take((size_t)BB * SS * HH * 2));
    bf16*  Wcat0  = (bf16*) (ws + take((size_t)4 * HH * (II + HH) * 2));
    bf16*  Wd0    = (bf16*) (ws + take((size_t)HH * HH * 2));
    bf16*  Wcat1  = (bf16*) (ws + take((size_t)4 * HH * (HH + HH) * 2));
    bf16*  Wd1    = (bf16*) (ws + take((size_t)HH * HH * 2));
    float* hbuf   = (float*)(ws + take((size_t)BB * HH * 4));
    float* cbuf   = (float*)(ws + take((size_t)BB * HH * 4));
    bf16*  hbf    = (bf16*) (ws + take((size_t)BB * HH * 2));
    bf16*  cbf    = (bf16*) (ws + take((size_t)BB * HH * 2));
    float* Gbuf   = (float*)(ws + take((size_t)BB * 4 * HH * 4));
    float* Dbuf   = (float*)(ws + take((size_t)BB * HH * 4));

    float* out_x = (float*)d_out;                       // [B,S,H]
    float* out_h = out_x + (size_t)BB * SS * HH;        // [L,B,H]
    float* out_c = out_h + (size_t)LL * BB * HH;        // [L,B,H]

    // ---- one-time prep ----
    compute_T_kernel<<<(BB * SS + 255) / 256, 256, 0, stream>>>(ts, Tbuf, (long)BB * SS);
    f32_to_bf16_kernel<<<2048, 256, 0, stream>>>(x_in, x0bf, (long)BB * SS * II);
    pack_gates_kernel<<<1024, 256, 0, stream>>>(P[0][pWi], P[0][pUi], P[0][pWf], P[0][pUf],
                                                P[0][pWo], P[0][pUo], P[0][pWc], P[0][pUc],
                                                Wcat0, II);
    pack_gates_kernel<<<1024, 256, 0, stream>>>(P[1][pWi], P[1][pUi], P[1][pWf], P[1][pUf],
                                                P[1][pWo], P[1][pUo], P[1][pWc], P[1][pUc],
                                                Wcat1, HH);
    f32_to_bf16_kernel<<<1024, 256, 0, stream>>>(P[0][pWd], Wd0, (long)HH * HH);
    f32_to_bf16_kernel<<<1024, 256, 0, stream>>>(P[1][pWd], Wd1, (long)HH * HH);

    for (int l = 0; l < LL; ++l) {
        hipMemsetAsync(hbuf, 0, (size_t)BB * HH * 4, stream);
        hipMemsetAsync(cbuf, 0, (size_t)BB * HH * 4, stream);
        hipMemsetAsync(hbf,  0, (size_t)BB * HH * 2, stream);
        hipMemsetAsync(cbf,  0, (size_t)BB * HH * 2, stream);

        const bf16* xin = (l == 0) ? x0bf : x1bf;
        const int   ind = (l == 0) ? II : HH;
        const long  ldx = (long)SS * ind;
        const bf16* Wc  = (l == 0) ? Wcat0 : Wcat1;
        const bf16* Wd  = (l == 0) ? Wd0 : Wd1;
        const int   Kg  = ind + HH;

        const dim3 gGates(4 * HH / TN, BB / TM);   // (32, 4)
        const dim3 gDecay(HH / TN, BB / TM);       // (8, 4)

        for (int t = 0; t < SS; ++t) {
            // gate pre-acts: [x_t | h] @ Wcat^T -> G [B, 4H]
            wmma_gemm_bf16<<<gGates, 128, 0, stream>>>(
                xin + (long)t * ind, ldx, hbf, (long)HH, ind,
                Wc, (long)Kg, Gbuf, 4L * HH, 4 * HH, Kg);
            // decay pre-act: c @ Wd^T -> Dp [B, H]
            wmma_gemm_bf16<<<gDecay, 128, 0, stream>>>(
                cbf, (long)HH, cbf, (long)HH, HH,
                Wd, (long)HH, Dbuf, (long)HH, HH, HH);
            // fused gates / state update
            float* outf = (l == LL - 1) ? (out_x + (long)t * HH) : nullptr;
            bf16*  outb = (l == 0)      ? (x1bf  + (long)t * HH) : nullptr;
            tlstm_point_kernel<<<BB * HH / 256, 256, 0, stream>>>(
                Gbuf, Dbuf, P[l][pbi], P[l][pbf], P[l][pbo], P[l][pbc], P[l][pbd],
                Tbuf + t, hbuf, cbuf, hbf, cbf, outf, outb);
        }
        hipMemcpyAsync(out_h + (size_t)l * BB * HH, hbuf, (size_t)BB * HH * 4,
                       hipMemcpyDeviceToDevice, stream);
        hipMemcpyAsync(out_c + (size_t)l * BB * HH, cbuf, (size_t)BB * HH * 4,
                       hipMemcpyDeviceToDevice, stream);
    }
}
